// RNNHeadAutoregressive_86535001080262
// MI455X (gfx1250) — compile-verified, matrix operand order checked by reading, not run on hardware
//
#include <hip/hip_runtime.h>
#include <hip/hip_bf16.h>

typedef __attribute__((ext_vector_type(16))) __bf16 v16bf;
typedef __attribute__((ext_vector_type(8)))  float  v8f;

#define HID    1024       // GRU hidden size (2*EMB)
#define NGATES 6144       // 3*HID (Wih) + 3*HID (Whh) concatenated
#define MROWS  256        // 2 spin-sequences * 128 batch stacked per GRU
#define BATCH  128
#define SEQ    256
#define BK     64
#define LDP    80         // padded LDS row stride (elements); 160B = 16B multiple

#if __has_builtin(__builtin_amdgcn_s_wait_asynccnt)
#define WAIT_ASYNC(n) __builtin_amdgcn_s_wait_asynccnt(n)
#else
#define WAIT_ASYNC(n) asm volatile("s_wait_asynccnt %0" ::"i"(n) : "memory")
#endif

// ---------------------------------------------------------------------------
// Prep kernels (run once per call; amortized over 256 steps)
// ---------------------------------------------------------------------------
__global__ void repack_weights(const float* __restrict__ Wih_up,
                               const float* __restrict__ Whh_up,
                               const float* __restrict__ Wih_dn,
                               const float* __restrict__ Whh_dn,
                               __bf16* __restrict__ Wbf) {
  size_t idx = (size_t)blockIdx.x * blockDim.x + threadIdx.x;
  size_t total = (size_t)2 * NGATES * HID;
  if (idx >= total) return;
  int g   = (int)(idx / ((size_t)NGATES * HID));
  size_t rem = idx - (size_t)g * NGATES * HID;
  int n = (int)(rem / HID);
  int k = (int)(rem % HID);
  float v;
  if (n < 3 * HID) {
    const float* W = (g == 0) ? Wih_up : Wih_dn;
    v = W[(size_t)n * HID + k];
  } else {
    const float* W = (g == 0) ? Whh_up : Whh_dn;
    v = W[(size_t)(n - 3 * HID) * HID + k];
  }
  Wbf[idx] = (__bf16)v;
}

__global__ void prep_bias(const float* __restrict__ bih_up,
                          const float* __restrict__ bhh_up,
                          const float* __restrict__ bih_dn,
                          const float* __restrict__ bhh_dn,
                          float* __restrict__ biasCat) {
  int idx = blockIdx.x * blockDim.x + threadIdx.x;
  if (idx >= 2 * NGATES) return;
  int g = idx / NGATES;
  int j = idx % NGATES;
  float v;
  if (j < 3 * HID) v = ((g == 0) ? bih_up : bih_dn)[j];
  else             v = ((g == 0) ? bhh_up : bhh_dn)[j - 3 * HID];
  biasCat[idx] = v;
}

// Builds X0[256][1024] = {[x, x] ; [x, -x]} in bf16 and zeroes Zero[2][256][1024].
__global__ void prep_x0(const float* __restrict__ x,
                        __bf16* __restrict__ X0,
                        __bf16* __restrict__ Zero) {
  int idx = blockIdx.x * blockDim.x + threadIdx.x;   // 0 .. 524287
  if (idx >= 2 * MROWS * HID) return;
  Zero[idx] = (__bf16)0.0f;
  if (idx < MROWS * HID) {
    int m = idx >> 10;          // row 0..255
    int j = idx & 1023;         // col 0..1023
    int b = m & 127;
    float v = x[(size_t)b * 512 + (j & 511)];
    if (j >= 512 && m >= 128) v = -v;   // spin_down second half negated
    X0[idx] = (__bf16)v;
  }
}

// ---------------------------------------------------------------------------
// Per-step gates GEMM:  G[g] = [ inp @ Wih_g^T | h @ Whh_g^T ]  -> [256, 6144]
// bf16 inputs, fp32 accumulate via v_wmma_f32_16x16x32_bf16.
// LDS staging uses GLOBAL_LOAD_ASYNC_TO_LDS_B128 (ASYNCcnt) with explicit
// 2-stage double buffering.
// grid: (48 N-tiles, 2 M-tiles, 2 GRUs), block 256 threads = 8 waves.
// ---------------------------------------------------------------------------
__global__ __launch_bounds__(256)
void gru_gates_gemm(const __bf16* __restrict__ inp_base, size_t inp_gstride,
                    const __bf16* __restrict__ h_base,   size_t h_gstride,
                    const __bf16* __restrict__ Wbf,
                    float* __restrict__ G) {
  __shared__ __align__(16) __bf16 Alds[2][128 * LDP];
  __shared__ __align__(16) __bf16 Blds[2][128 * LDP];

  const int n0 = blockIdx.x * 128;
  const int m0 = blockIdx.y * 128;
  const int g  = blockIdx.z;

  const __bf16* A = (n0 < 3 * HID) ? (inp_base + (size_t)g * inp_gstride)
                                   : (h_base   + (size_t)g * h_gstride);
  const __bf16* W = Wbf + ((size_t)g * NGATES + n0) * HID;
  float* Gg = G + (size_t)g * MROWS * NGATES;

  const unsigned long long Abase = (unsigned long long)(size_t)A;
  const unsigned long long Wbase = (unsigned long long)(size_t)W;

  const int tid  = threadIdx.x;
  const int lane = tid & 31;
  const int wid  = tid >> 5;
  const int wm0  = (wid & 3) * 32;   // 4 waves along M (32 rows each)
  const int wn0  = (wid >> 2) * 64;  // 2 waves along N (64 cols each)
  const int half = lane >> 4;        // lane group 0/1
  const int r    = lane & 15;

  // This thread's staging slot: 4 chunks of 8 bf16 per array per tile.
  v8f acc[2][4];
#pragma unroll
  for (int i = 0; i < 2; i++)
#pragma unroll
    for (int j = 0; j < 4; j++)
      acc[i][j] = (v8f){0.f, 0.f, 0.f, 0.f, 0.f, 0.f, 0.f, 0.f};

  // Issue one 128x64 A tile + one 128x64 W tile into LDS stage `st`
  // via per-lane async b128 copies (8 async ops per thread).
  auto issue_tile = [&](int k0, int st) {
#pragma unroll
    for (int rr = 0; rr < 4; rr++) {
      int c   = rr * 256 + tid;        // 0..1023 chunk id
      int row = c >> 3;
      int col = (c & 7) * 8;
      unsigned ldsA  = (unsigned)(size_t)&Alds[st][row * LDP + col];
      unsigned voffA = (unsigned)(((m0 + row) * HID + k0 + col) * 2);
      asm volatile("global_load_async_to_lds_b128 %0, %1, %2"
                   :: "v"(ldsA), "v"(voffA), "s"(Abase) : "memory");
      unsigned ldsB  = (unsigned)(size_t)&Blds[st][row * LDP + col];
      unsigned voffB = (unsigned)((row * HID + k0 + col) * 2);
      asm volatile("global_load_async_to_lds_b128 %0, %1, %2"
                   :: "v"(ldsB), "v"(voffB), "s"(Wbase) : "memory");
    }
  };

  int stage = 0;
  issue_tile(0, 0);

  for (int k0 = 0; k0 < HID; k0 += BK) {
    const bool has_next = (k0 + BK) < HID;
    if (has_next) {
      issue_tile(k0 + BK, stage ^ 1);   // prefetch next tile into other stage
      WAIT_ASYNC(8);                    // 8 oldest = current stage complete
    } else {
      WAIT_ASYNC(0);
    }
    __syncthreads();                    // all waves' async data visible

    const __bf16* Acur = Alds[stage];
    const __bf16* Bcur = Blds[stage];
#pragma unroll
    for (int kk = 0; kk < BK; kk += 32) {
      // 16-bit operand layout: lanes 0-15 hold K 0..7 (VGPR0-3) and 16..23
      // (VGPR4-7); lanes 16-31 hold K 8..15 and 24..31.
      const int klo = kk + half * 8;
      const int khi = kk + 16 + half * 8;
      v16bf afr[2], bfr[4];
#pragma unroll
      for (int ti = 0; ti < 2; ti++) {
        int row = wm0 + ti * 16 + r;
        uint4* p = (uint4*)&afr[ti];
        p[0] = *(const uint4*)&Acur[row * LDP + klo];
        p[1] = *(const uint4*)&Acur[row * LDP + khi];
      }
#pragma unroll
      for (int tj = 0; tj < 4; tj++) {
        int row = wn0 + tj * 16 + r;   // W stored [N][K] -> mirrors A pattern
        uint4* p = (uint4*)&bfr[tj];
        p[0] = *(const uint4*)&Bcur[row * LDP + klo];
        p[1] = *(const uint4*)&Bcur[row * LDP + khi];
      }
#pragma unroll
      for (int ti = 0; ti < 2; ti++)
#pragma unroll
        for (int tj = 0; tj < 4; tj++)
          acc[ti][tj] = __builtin_amdgcn_wmma_f32_16x16x32_bf16(
              false, afr[ti], false, bfr[tj], (short)0, acc[ti][tj],
              false, false);
    }
    __syncthreads();   // everyone done reading `stage` before it is re-filled
    stage ^= 1;
  }

  // Epilogue: C/D layout — VGPR i: lanes 0-15 -> M=i, lanes 16-31 -> M=8+i;
  // N = lane&15 within the tile.
#pragma unroll
  for (int ti = 0; ti < 2; ti++)
#pragma unroll
    for (int tj = 0; tj < 4; tj++) {
      int gcol = n0 + wn0 + tj * 16 + r;
      int grow = m0 + wm0 + ti * 16 + half * 8;
#pragma unroll
      for (int i = 0; i < 8; i++)
        Gg[(size_t)(grow + i) * NGATES + gcol] = acc[ti][tj][i];
    }
}

// ---------------------------------------------------------------------------
// Per-step gate combine: r/z/n gating, writes bf16 state + fp32 output slice.
// ---------------------------------------------------------------------------
__global__ __launch_bounds__(256)
void gru_gate_combine(const float* __restrict__ G,
                      const float* __restrict__ biasCat,
                      const __bf16* __restrict__ Hprev,
                      __bf16* __restrict__ Hout,
                      float* __restrict__ out, int t) {
  int idx = blockIdx.x * blockDim.x + threadIdx.x;   // 0 .. 2*256*1024-1
  if (idx >= 2 * MROWS * HID) return;
  int g   = idx >> 18;           // 256*1024 = 2^18
  int rem = idx & 262143;
  int m   = rem >> 10;
  int j   = rem & 1023;

  const float* Gg = G + (size_t)g * MROWS * NGATES + (size_t)m * NGATES;
  const float* bb = biasCat + (size_t)g * NGATES;

  float ir  = Gg[j]         + bb[j];
  float iz  = Gg[1024 + j]  + bb[1024 + j];
  float in_ = Gg[2048 + j]  + bb[2048 + j];
  float hr  = Gg[3072 + j]  + bb[3072 + j];
  float hz  = Gg[4096 + j]  + bb[4096 + j];
  float hn  = Gg[5120 + j]  + bb[5120 + j];

  float rg = 1.f / (1.f + __expf(-(ir + hr)));
  float zg = 1.f / (1.f + __expf(-(iz + hz)));
  float ng = tanhf(in_ + rg * hn);

  size_t hoff = (size_t)g * MROWS * HID + (size_t)m * HID + j;
  float hp = (float)Hprev[hoff];
  float hnew = (1.f - zg) * ng + zg * hp;

  Hout[hoff] = (__bf16)hnew;

  int seq = g * 2 + (m >> 7);    // (up_su, up_sd, dn_su, dn_sd)
  int b   = m & 127;
  out[((size_t)seq * BATCH + b) * (size_t)(SEQ * HID) + (size_t)t * HID + j] =
      hnew;
}

// ---------------------------------------------------------------------------
extern "C" void kernel_launch(void* const* d_in, const int* in_sizes, int n_in,
                              void* d_out, int out_size, void* d_ws,
                              size_t ws_size, hipStream_t stream) {
  (void)in_sizes; (void)n_in; (void)out_size; (void)ws_size;
  const float* x      = (const float*)d_in[0];
  const float* Wih_up = (const float*)d_in[1];
  const float* Whh_up = (const float*)d_in[2];
  const float* Wih_dn = (const float*)d_in[3];
  const float* Whh_dn = (const float*)d_in[4];
  const float* bih_up = (const float*)d_in[5];
  const float* bhh_up = (const float*)d_in[6];
  const float* bih_dn = (const float*)d_in[7];
  const float* bhh_dn = (const float*)d_in[8];
  float* out = (float*)d_out;

  char* ws = (char*)d_ws;
  size_t o = 0;
  __bf16* Wbf  = (__bf16*)(ws + o); o += (size_t)2 * NGATES * HID * 2;  // 25.2 MB
  float*  bias = (float*) (ws + o); o += (size_t)2 * NGATES * 4;        // 48 KB
  __bf16* X0   = (__bf16*)(ws + o); o += (size_t)MROWS * HID * 2;       // 512 KB
  __bf16* Zero = (__bf16*)(ws + o); o += (size_t)2 * MROWS * HID * 2;   // 1 MB
  __bf16* H    = (__bf16*)(ws + o); o += (size_t)2 * MROWS * HID * 2;   // 1 MB
  float*  G    = (float*) (ws + o);                                     // 12.6 MB

  // --- one-time prep ---
  {
    size_t tot = (size_t)2 * NGATES * HID;
    repack_weights<<<(unsigned)((tot + 255) / 256), 256, 0, stream>>>(
        Wih_up, Whh_up, Wih_dn, Whh_dn, Wbf);
    prep_bias<<<(2 * NGATES + 255) / 256, 256, 0, stream>>>(
        bih_up, bhh_up, bih_dn, bhh_dn, bias);
    prep_x0<<<(2 * MROWS * HID + 255) / 256, 256, 0, stream>>>(x, X0, Zero);
  }

  const size_t gstride = (size_t)MROWS * HID;
  for (int t = 0; t < SEQ; t++) {
    const __bf16* inp    = (t == 0) ? X0 : H;
    size_t        inp_gs = (t == 0) ? 0 : gstride;   // X0 shared by both GRUs
    const __bf16* hb     = (t == 0) ? Zero : H;
    gru_gates_gemm<<<dim3(48, 2, 2), 256, 0, stream>>>(
        inp, inp_gs, hb, gstride, Wbf, G);
    const __bf16* hprev = (t == 0) ? Zero : H;
    gru_gate_combine<<<(2 * MROWS * HID) / 256, 256, 0, stream>>>(
        G, bias, hprev, H, out, t);
  }
}
